// MixedQuantLinear_44521630991150
// MI455X (gfx1250) — compile-verified, matrix operand order checked by reading, not run on hardware
//
#include <hip/hip_runtime.h>
#include <hip/hip_bf16.h>
#include <cstdint>

typedef __attribute__((ext_vector_type(16))) _Float16 v16h;
typedef __attribute__((ext_vector_type(8)))  _Float16 v8h;
typedef __attribute__((ext_vector_type(8)))  float    v8f;

#define IN_FEATURES  4096
#define OUT_FEATURES 11008
#define TOKENS       64

#define KSTEP   128                       // K elements staged per pipeline stage
#define DEPTH   4                         // LDS pipeline depth (quad buffer)
#define NSTAGES (IN_FEATURES / KSTEP)     // 32
#define TILE_BYTES (16 * KSTEP * 2)       // 4 KB per stage

// ---------------------------------------------------------------------------
// Kernel 1: dequantize/scatter compacted mixed-quant weights into f16 W (d_ws).
// int8_pos and fp16_pos partition [0, numel), so W is fully initialized.
// ---------------------------------------------------------------------------
__global__ __launch_bounds__(256) void dequant_scatter_kernel(
    const int8_t*   __restrict__ int8_data,
    const _Float16* __restrict__ fp16_data,
    const _Float16* __restrict__ scales,
    const int*      __restrict__ int8_pos,
    const int*      __restrict__ fp16_pos,
    const int*      __restrict__ block_idx,
    _Float16*       __restrict__ W,
    int n_int8, int n_fp16)
{
    const size_t stride = (size_t)gridDim.x * blockDim.x;
    for (size_t i = (size_t)blockIdx.x * blockDim.x + threadIdx.x;
         i < (size_t)n_int8; i += stride) {
        _Float16 q = (_Float16)(float)int8_data[i];
        W[(size_t)(uint32_t)int8_pos[i]] = q * scales[(uint32_t)block_idx[i]];
    }
    for (size_t i = (size_t)blockIdx.x * blockDim.x + threadIdx.x;
         i < (size_t)n_fp16; i += stride) {
        W[(size_t)(uint32_t)fp16_pos[i]] = fp16_data[i];
    }
}

// Async global->LDS 16-byte copy (ASYNCcnt DMA path, gfx1250).
__device__ __forceinline__ void async_copy_b128(uint32_t lds_off, const void* gaddr) {
    asm volatile("global_load_async_to_lds_b128 %0, %1, off"
                 :: "v"(lds_off), "v"((unsigned long long)(uintptr_t)gaddr)
                 : "memory");
}
__device__ __forceinline__ void wait_async_le4() { asm volatile("s_wait_asynccnt 0x4" ::: "memory"); }
__device__ __forceinline__ void wait_async_le2() { asm volatile("s_wait_asynccnt 0x2" ::: "memory"); }
__device__ __forceinline__ void wait_async_le0() { asm volatile("s_wait_asynccnt 0x0" ::: "memory"); }

// Consume one staged W tile: 4 x (K=32) WMMA steps for this wave's M tile.
__device__ __forceinline__ void consume_stage(
    const _Float16* __restrict__ wt,    // LDS stage base
    const _Float16* __restrict__ arow,  // this lane's x row
    int k0, int laneR, int ksel, v8f& acc)
{
    #pragma unroll
    for (int kk = 0; kk < KSTEP; kk += 32) {
        // W fragment from LDS: lane holds row laneR, K = kk+ksel..+8, +16..+24
        v8h b_lo = *(const v8h*)(wt + laneR * KSTEP + kk + ksel);
        v8h b_hi = *(const v8h*)(wt + laneR * KSTEP + kk + ksel + 16);
        v16h bfrag = __builtin_shufflevector(b_lo, b_hi,
            0,1,2,3,4,5,6,7, 8,9,10,11,12,13,14,15);

        // A fragment from global (x is L2/WGP$-resident)
        v8h a_lo = *(const v8h*)(arow + k0 + kk + ksel);
        v8h a_hi = *(const v8h*)(arow + k0 + kk + ksel + 16);
        v16h afrag = __builtin_shufflevector(a_lo, a_hi,
            0,1,2,3,4,5,6,7, 8,9,10,11,12,13,14,15);

        acc = __builtin_amdgcn_wmma_f32_16x16x32_f16(
            false, afrag, false, bfrag, (short)0, acc, false, false);
    }
}

// ---------------------------------------------------------------------------
// Kernel 2: WMMA GEMM  out[m,n] = sum_k x[m,k]*W[n,k] + bias[n]
// Block = 128 threads (4 waves) owns one 16-wide N tile; wave w owns M tile w.
// W stream is staged through a quad-buffered LDS pipeline via async copies
// (shared by all 4 waves); x fragments are direct global loads (cache hits).
// Steady-state loop is branchless; the 3-stage pipeline drain is peeled.
// ---------------------------------------------------------------------------
__global__ __launch_bounds__(128) void wmma_gemm_kernel(
    const _Float16* __restrict__ x,     // (64, 4096)
    const _Float16* __restrict__ W,     // (11008, 4096) dequantized
    const _Float16* __restrict__ bias,  // (11008,)
    float*          __restrict__ out)   // (64, 11008)
{
    __shared__ __align__(16) _Float16 wtile[DEPTH][16 * KSTEP];  // 16 KB

    const int t     = threadIdx.x;
    const int lane  = t & 31;
    const int wave  = t >> 5;              // M tile index 0..3
    const int n0    = blockIdx.x * 16;
    const int laneR = lane & 15;           // matrix row this lane owns
    const int ksel  = (lane >> 4) * 8;     // K sub-offset: 0 or 8

    const uint32_t lds_base = (uint32_t)(uintptr_t)&wtile[0][0];

    // Async-copy mapping: tile = 16 rows x 256 B; chunk c (0..255) is 16 B.
    // Thread t copies chunks t and t+128.
    const int      row0 = t >> 4,            row1 = (t + 128) >> 4;
    const int      cb0  = (t & 15) * 16,     cb1  = ((t + 128) & 15) * 16;
    const char*    wrow0 = (const char*)(W + (size_t)(n0 + row0) * IN_FEATURES) + cb0;
    const char*    wrow1 = (const char*)(W + (size_t)(n0 + row1) * IN_FEATURES) + cb1;
    const uint32_t ldsc0 = lds_base + (uint32_t)t * 16;
    const uint32_t ldsc1 = lds_base + (uint32_t)(t + 128) * 16;

    #define ISSUE_STAGE(s)                                                        \
        do {                                                                      \
            uint32_t soff = (uint32_t)((s) & (DEPTH - 1)) * TILE_BYTES;           \
            size_t   goff = (size_t)(s) * (KSTEP * 2);                            \
            async_copy_b128(ldsc0 + soff, wrow0 + goff);                          \
            async_copy_b128(ldsc1 + soff, wrow1 + goff);                          \
        } while (0)

    // Prologue: fill stages 0..DEPTH-2.
    ISSUE_STAGE(0);
    ISSUE_STAGE(1);
    ISSUE_STAGE(2);

    const _Float16* arow = x + (size_t)(wave * 16 + laneR) * IN_FEATURES;
    v8f acc = v8f{};

    // Steady state: stage s is resident once <= 2*(DEPTH-2) async ops remain
    // (per-wave async completion is in-order); barrier makes all waves' copies
    // visible; then issue stage s+3 and consume stage s.
    for (int s = 0; s < NSTAGES - (DEPTH - 1); ++s) {
        wait_async_le4();
        __syncthreads();
        ISSUE_STAGE(s + DEPTH - 1);
        consume_stage(&wtile[s & (DEPTH - 1)][0], arow, s * KSTEP, laneR, ksel, acc);
    }
    #undef ISSUE_STAGE

    // Pipeline drain: last 3 stages, no new issues.
    {
        const int s0 = NSTAGES - 3, s1 = NSTAGES - 2, s2 = NSTAGES - 1;
        wait_async_le4();
        __syncthreads();
        consume_stage(&wtile[s0 & (DEPTH - 1)][0], arow, s0 * KSTEP, laneR, ksel, acc);
        wait_async_le2();
        __syncthreads();
        consume_stage(&wtile[s1 & (DEPTH - 1)][0], arow, s1 * KSTEP, laneR, ksel, acc);
        wait_async_le0();
        __syncthreads();
        consume_stage(&wtile[s2 & (DEPTH - 1)][0], arow, s2 * KSTEP, laneR, ksel, acc);
    }

    // C/D layout: VGPR r, lanes 0..15 -> (M=r, N=lane); lanes 16..31 -> (M=r+8).
    const int   n  = n0 + laneR;
    const int   mh = (lane >> 4) * 8;
    const float bv = (float)bias[n];

    #pragma unroll
    for (int r = 0; r < 8; ++r) {
        int m = wave * 16 + mh + r;
        out[(size_t)m * OUT_FEATURES + n] = acc[r] + bv;
    }
}

extern "C" void kernel_launch(void* const* d_in, const int* in_sizes, int n_in,
                              void* d_out, int out_size, void* d_ws, size_t ws_size,
                              hipStream_t stream) {
    const _Float16* x         = (const _Float16*)d_in[0];
    const int8_t*   int8_data = (const int8_t*)  d_in[1];
    const _Float16* fp16_data = (const _Float16*)d_in[2];
    const _Float16* scales    = (const _Float16*)d_in[3];
    const _Float16* bias      = (const _Float16*)d_in[4];
    const int*      int8_pos  = (const int*)     d_in[5];
    const int*      fp16_pos  = (const int*)     d_in[6];
    const int*      block_idx = (const int*)     d_in[7];

    const int n_int8 = in_sizes[1];
    const int n_fp16 = in_sizes[2];

    _Float16* W = (_Float16*)d_ws;   // 11008*4096*2B = ~90.2 MB scratch
    float*    out = (float*)d_out;

    // Stage 1: scatter-dequantize W.
    dequant_scatter_kernel<<<4096, 256, 0, stream>>>(
        int8_data, fp16_data, scales, int8_pos, fp16_pos, block_idx,
        W, n_int8, n_fp16);

    // Stage 2: WMMA GEMM, one block per 16-wide N tile (688 blocks x 4 waves).
    wmma_gemm_kernel<<<OUT_FEATURES / 16, 128, 0, stream>>>(x, W, bias, out);
}
